// CandidateFinder_78340203479204
// MI455X (gfx1250) — 8-bit matrix operand order repaired and checked by reading, compiled, not run on hardware
//
#include <hip/hip_runtime.h>
#include <hip/hip_bf16.h>
#include <stdint.h>

#define BATCH   2
#define LEN     4096
#define DIM     64
#define KMAX    64
#define TILES_PER_BATCH (LEN / 16)
#define N_TILES (BATCH * TILES_PER_BATCH)
#define TILE_BYTES 1024

typedef __attribute__((ext_vector_type(8))) int v8i;

__global__ void pack_q_kernel(const float* __restrict__ q,
                              unsigned* __restrict__ wsq) {
    int tid  = blockIdx.x * blockDim.x + threadIdx.x;
    int tile = tid >> 5;
    int lane = tid & 31;
    int b    = tile >> 8;
    int row  = ((tile & 255) << 4) + (lane & 15);
    int hi   = lane >> 4;
    const float* src = q + ((size_t)(b * LEN + row)) * DIM;

    unsigned w[8];
#pragma unroll
    for (int c = 0; c < 4; ++c) {
        int K0 = (c >> 1) * 32 + hi * 16 + (c & 1) * 8;
        unsigned lo = 0, hw = 0;
#pragma unroll
        for (int t = 0; t < 4; ++t) {
            lo |= ((src[K0 + t]     > 0.0f) ? 0x01u : 0xFFu) << (8 * t);
            hw |= ((src[K0 + 4 + t] > 0.0f) ? 0x01u : 0xFFu) << (8 * t);
        }
        w[c * 2]     = lo;
        w[c * 2 + 1] = hw;
    }
    uint4* dst = (uint4*)((char*)wsq + (size_t)tile * TILE_BYTES + lane * 32);
    dst[0] = make_uint4(w[0], w[1], w[2], w[3]);
    dst[1] = make_uint4(w[4], w[5], w[6], w[7]);
}

__global__ void pack_k_kernel(const float* __restrict__ k,
                              unsigned* __restrict__ wsk) {
    int tid  = blockIdx.x * blockDim.x + threadIdx.x;
    int tile = tid >> 5;
    int lane = tid & 31;
    int b    = tile >> 8;
    int col  = ((tile & 255) << 4) + (lane & 15);
    int g    = lane >> 4;
    const float* src = k + ((size_t)(b * LEN + col)) * DIM;

    unsigned w[8];
#pragma unroll
    for (int c = 0; c < 2; ++c) {
        int K0 = c * 32 + g * 16;
#pragma unroll
        for (int d = 0; d < 4; ++d) {
            unsigned acc = 0;
#pragma unroll
            for (int t = 0; t < 4; ++t)
                acc |= ((src[K0 + d * 4 + t] > 0.0f) ? 0x01u : 0xFFu) << (8 * t);
            w[c * 4 + d] = acc;
        }
    }
    uint4* dst = (uint4*)((char*)wsk + (size_t)tile * TILE_BYTES + lane * 32);
    dst[0] = make_uint4(w[0], w[1], w[2], w[3]);
    dst[1] = make_uint4(w[4], w[5], w[6], w[7]);
}

__global__ void match_kernel(const char* __restrict__ wsq,
                             const char* __restrict__ wsk,
                             float* __restrict__ out) {
    int wave = (blockIdx.x * blockDim.x + threadIdx.x) >> 5;
    int lane = threadIdx.x & 31;
    int b    = wave >> 8;
    int it   = wave & 255;

    v8i A = *(const v8i*)(wsq + (size_t)wave * TILE_BYTES + lane * 32);

    const char* kbase = wsk + (size_t)b * TILES_PER_BATCH * TILE_BYTES;
    float* outrow0 = out + ((size_t)(b * LEN + it * 16)) * KMAX;

    int half8   = (lane >> 4) << 3;
    unsigned halfmask = 0xFFFFu << ((lane >> 4) << 4);
    unsigned lowmask  = (1u << lane) - 1u;

    int cnt[8];
#pragma unroll
    for (int r = 0; r < 8; ++r) cnt[r] = 0;

#pragma unroll 2
    for (int jt = 0; jt < TILES_PER_BATCH; ++jt) {
        v8i Bm = *(const v8i*)(kbase + (size_t)jt * TILE_BYTES + lane * 32);
        v8i C  = {0, 0, 0, 0, 0, 0, 0, 0};
        v8i D = __builtin_amdgcn_wmma_i32_16x16x64_iu8(
             true, A,  true, Bm, C,
             false,  false);

        bool any = false;
#pragma unroll
        for (int r = 0; r < 8; ++r) any |= (D[r] == 64);
        if (__builtin_amdgcn_ballot_w32(any) != 0u) {
            float jf = (float)(jt * 16 + (lane & 15));
#pragma unroll
            for (int r = 0; r < 8; ++r) {
                bool hit = (D[r] == 64);
                unsigned m = __builtin_amdgcn_ballot_w32(hit);
                int inc = __builtin_popcount(m & halfmask);
                if (hit) {
                    int pos = cnt[r] + __builtin_popcount(m & halfmask & lowmask);
                    if (pos < KMAX)
                        outrow0[(size_t)(r + half8) * KMAX + pos] = jf;
                }
                cnt[r] += inc;
            }
        }
    }

#pragma unroll
    for (int r = 0; r < 8; ++r) {
        int row = r + half8;
        for (int c = (lane & 15); c < KMAX; c += 16)
            if (c >= cnt[r])
                outrow0[(size_t)row * KMAX + c] = -1.0f;
    }
}

extern "C" void kernel_launch(void* const* d_in, const int* in_sizes, int n_in,
                              void* d_out, int out_size, void* d_ws, size_t ws_size,
                              hipStream_t stream) {
    const float* q = (const float*)d_in[0];
    const float* k = (const float*)d_in[1];
    float* out = (float*)d_out;

    char* wsq = (char*)d_ws;
    char* wsk = wsq + (size_t)N_TILES * TILE_BYTES;

    pack_q_kernel<<<(N_TILES * 32) / 256, 256, 0, stream>>>(q, (unsigned*)wsq);
    pack_k_kernel<<<(N_TILES * 32) / 256, 256, 0, stream>>>(k, (unsigned*)wsk);

    match_kernel<<<N_TILES / 4, 128, 0, stream>>>(wsq, wsk, out);
}
